// TransformerEncoderLayer_57827439673577
// MI455X (gfx1250) — matrix operand order repaired and checked by reading, not compiled, not run on hardware
//
#include <hip/hip_runtime.h>

typedef _Float16 hf;
typedef _Float16 v16h __attribute__((ext_vector_type(16)));
typedef _Float16 h8   __attribute__((ext_vector_type(8)));
typedef _Float16 h4   __attribute__((ext_vector_type(4)));
typedef float    v8f  __attribute__((ext_vector_type(8)));
typedef int      v4i  __attribute__((ext_vector_type(4)));

// ---------------------------------------------------------------------------
// CDNA5 async global->LDS copy (ASYNCcnt path), with fallback.
// Builtin signature (from hipcc diagnostic): first arg is v4i* (global),
// second v4i* (LDS), then imm offset, imm cpol.
// ---------------------------------------------------------------------------
#if defined(__has_builtin)
#if __has_builtin(__builtin_amdgcn_global_load_async_to_lds_b128)
#define HAVE_ASYNC_LDS 1
#endif
#endif
#ifndef HAVE_ASYNC_LDS
#define HAVE_ASYNC_LDS 0
#endif

#if HAVE_ASYNC_LDS
typedef __attribute__((address_space(1))) v4i* gptr_v4i;
typedef __attribute__((address_space(3))) v4i* lptr_v4i;

__device__ __forceinline__ void async_b128(const hf* g, hf* l)
{
    __builtin_amdgcn_global_load_async_to_lds_b128(
        (gptr_v4i)(g), (lptr_v4i)(l), 0, 0);
}
__device__ __forceinline__ void wait_async0()
{
#if __has_builtin(__builtin_amdgcn_s_wait_asynccnt)
    __builtin_amdgcn_s_wait_asynccnt(0);
#else
    asm volatile("s_wait_asynccnt 0x0" ::: "memory");
#endif
}
#endif

// ---------------------------------------------------------------------------
// WMMA fragment helpers (CDNA5 16x16x32 f16, wave32)
// A (16x32, MxK): lanes 0-15 rows M=0..15 hold K = {0..7, 16..23};
//                 lanes 16-31 rows M=0..15 hold K = {8..15, 24..31}.
// B (32x16, KxN): lane = N (mod 16); lanes 0-15 hold K = {0..7, 16..23}, lanes 16-31 K = {8..15, 24..31}
//                 (same as A; source stored "transposed": row = n, cols = k).
// C/D (16x16):    vgpr r, lane l -> M = r + 8*(l>=16), N = l&15.
// ---------------------------------------------------------------------------
__device__ __forceinline__ v16h load_a_frag(const hf* base, int ld, int row0, int col0, int lane)
{
    int m  = lane & 15;
    int hi = (lane >> 4) & 1;
    const hf* p = base + (size_t)(row0 + m) * ld + col0 + hi * 8;
    h8 lo  = *(const h8*)p;
    h8 hi8 = *(const h8*)(p + 16);
    v16h r;
#pragma unroll
    for (int i = 0; i < 8; ++i) { r[i] = lo[i]; r[i + 8] = hi8[i]; }
    return r;
}

__device__ __forceinline__ v16h load_bt_frag(const hf* base, int ld, int n0, int k0, int lane)
{
    int n  = lane & 15;
    int hi = (lane >> 4) & 1;
    const hf* p = base + (size_t)(n0 + n) * ld + k0 + hi * 8;
    h8 a = *(const h8*)p;
    h8 b = *(const h8*)(p + 16);
    v16h r;
#pragma unroll
    for (int i = 0; i < 8; ++i) { r[i] = a[i]; r[i + 8] = b[i]; }
    return r;
}

__device__ __forceinline__ v8f wmma16(v16h a, v16h b, v8f c)
{
    return __builtin_amdgcn_wmma_f32_16x16x32_f16(false, a, false, b, (short)0, c, false, false);
}

// ---------------------------------------------------------------------------
// Weight transpose + f32 -> f16 cast:  Wt[n][k] = (hf)W[k][n]
// ---------------------------------------------------------------------------
__global__ __launch_bounds__(256) void transpose_cast_f16(const float* __restrict__ W,
                                                          hf* __restrict__ Wt, int K, int N)
{
    __shared__ float tile[32][33];
    int n0 = blockIdx.x * 32;
    int k0 = blockIdx.y * 32;
    int tx = threadIdx.x;           // 0..31
    for (int r = threadIdx.y; r < 32; r += 8)
        tile[r][tx] = W[(size_t)(k0 + r) * N + n0 + tx];
    __syncthreads();
    for (int r = threadIdx.y; r < 32; r += 8)
        Wt[(size_t)(n0 + r) * K + k0 + tx] = (hf)tile[tx][r];
}

// ---------------------------------------------------------------------------
// LayerNorm (row = 1024 f32) -> f16 output.  One block (256 thr) per row.
// ---------------------------------------------------------------------------
__global__ __launch_bounds__(256) void layernorm_f16(const float* __restrict__ x,
                                                     const float* __restrict__ g,
                                                     const float* __restrict__ bta,
                                                     hf* __restrict__ out, int D)
{
    int row = blockIdx.x;
    int t = threadIdx.x, lane = t & 31, wave = t >> 5;
    const float* xr = x + (size_t)row * D;
    int i0 = t * 4;
    float4 xv = *(const float4*)(xr + i0);
    float s  = xv.x + xv.y + xv.z + xv.w;
    float s2 = xv.x * xv.x + xv.y * xv.y + xv.z * xv.z + xv.w * xv.w;
#pragma unroll
    for (int off = 16; off > 0; off >>= 1) {
        s  += __shfl_xor(s,  off, 32);
        s2 += __shfl_xor(s2, off, 32);
    }
    __shared__ float ps[8], ps2[8];
    if (lane == 0) { ps[wave] = s; ps2[wave] = s2; }
    __syncthreads();
    float ts = 0.f, ts2 = 0.f;
#pragma unroll
    for (int w = 0; w < 8; ++w) { ts += ps[w]; ts2 += ps2[w]; }
    float mu  = ts / (float)D;
    float var = ts2 / (float)D - mu * mu;
    float inv = rsqrtf(var + 1e-5f);
    float4 gv = *(const float4*)(g + i0);
    float4 bv = *(const float4*)(bta + i0);
    h4 o;
    o[0] = (hf)((xv.x - mu) * inv * gv.x + bv.x);
    o[1] = (hf)((xv.y - mu) * inv * gv.y + bv.y);
    o[2] = (hf)((xv.z - mu) * inv * gv.z + bv.z);
    o[3] = (hf)((xv.w - mu) * inv * gv.w + bv.w);
    *(h4*)(out + (size_t)row * D + i0) = o;
}

// ---------------------------------------------------------------------------
// GEMM: C[M,N] = A[M,K](f16) * Bt[N,K](f16)^T  (+bias)(+residual)(relu)
// 128x128 macro tile, 8 waves, wave = 64x32 sub-tile.
// Async path: K-step 64, double-buffered LDS, global_load_async_to_lds_b128.
// ---------------------------------------------------------------------------
__global__ __launch_bounds__(256) void gemm_f16(const hf* __restrict__ A,
                                                const hf* __restrict__ Bt,
                                                int M, int N, int K,
                                                const float* __restrict__ bias,
                                                const float* __restrict__ residual,
                                                int relu,
                                                float* __restrict__ outF,
                                                hf* __restrict__ outH)
{
    int t = threadIdx.x, lane = t & 31, wave = t >> 5;
    int mBase = blockIdx.y * 128;
    int nBase = blockIdx.x * 128;
    int mW = (wave >> 2) * 64;      // 0 or 64
    int nW = (wave & 3) * 32;       // 0,32,64,96

    v8f acc[4][2] = {};

#if HAVE_ASYNC_LDS
    __shared__ hf As[2][128][72];
    __shared__ hf Bs[2][128][72];

    int lr = t >> 1;                // 0..127
    int lc = (t & 1) * 32;          // 0 or 32 (x 64B)
    const hf* pa0 = A  + (size_t)(mBase + lr) * K + lc;
    const hf* pb0 = Bt + (size_t)(nBase + lr) * K + lc;

    auto issue = [&](int buf, int k0) {
        const hf* pa = pa0 + k0;
        const hf* pb = pb0 + k0;
        hf* la = &As[buf][lr][lc];
        hf* lb = &Bs[buf][lr][lc];
#pragma unroll
        for (int u = 0; u < 4; ++u) {
            async_b128(pa + u * 8, la + u * 8);
            async_b128(pb + u * 8, lb + u * 8);
        }
    };

    issue(0, 0);
    int nIter = K >> 6;             // K-step 64
    for (int it = 0; it < nIter; ++it) {
        int buf = it & 1;
        wait_async0();              // this wave's loads into 'buf' landed
        __syncthreads();            // everyone's loads landed / prev reads done
        if (it + 1 < nIter) issue(buf ^ 1, (it + 1) * 64);
#pragma unroll
        for (int kk = 0; kk < 2; ++kk) {
            v16h bf0 = load_bt_frag(&Bs[buf][0][0], 72, nW,      kk * 32, lane);
            v16h bf1 = load_bt_frag(&Bs[buf][0][0], 72, nW + 16, kk * 32, lane);
#pragma unroll
            for (int i = 0; i < 4; ++i) {
                v16h af = load_a_frag(&As[buf][0][0], 72, mW + 16 * i, kk * 32, lane);
                acc[i][0] = wmma16(af, bf0, acc[i][0]);
                acc[i][1] = wmma16(af, bf1, acc[i][1]);
            }
        }
    }
#else
    __shared__ hf As[128][40];
    __shared__ hf Bs[128][40];
    int lr = t >> 1;                // 0..127
    int lc = (t & 1) * 16;          // 0 or 16
    for (int k0 = 0; k0 < K; k0 += 32) {
        const hf* pa = A  + (size_t)(mBase + lr) * K + k0 + lc;
        const hf* pb = Bt + (size_t)(nBase + lr) * K + k0 + lc;
        h8 a0 = *(const h8*)pa;       h8 a1 = *(const h8*)(pa + 8);
        h8 b0 = *(const h8*)pb;       h8 b1 = *(const h8*)(pb + 8);
        __syncthreads();
        *(h8*)&As[lr][lc]     = a0;   *(h8*)&As[lr][lc + 8] = a1;
        *(h8*)&Bs[lr][lc]     = b0;   *(h8*)&Bs[lr][lc + 8] = b1;
        __syncthreads();
        v16h bf0 = load_bt_frag(&Bs[0][0], 40, nW,      0, lane);
        v16h bf1 = load_bt_frag(&Bs[0][0], 40, nW + 16, 0, lane);
#pragma unroll
        for (int i = 0; i < 4; ++i) {
            v16h af = load_a_frag(&As[0][0], 40, mW + 16 * i, 0, lane);
            acc[i][0] = wmma16(af, bf0, acc[i][0]);
            acc[i][1] = wmma16(af, bf1, acc[i][1]);
        }
    }
#endif

    int hi = lane >> 4, nl = lane & 15;
#pragma unroll
    for (int i = 0; i < 4; ++i)
#pragma unroll
        for (int j = 0; j < 2; ++j)
#pragma unroll
            for (int r = 0; r < 8; ++r) {
                int row = mBase + mW + 16 * i + r + hi * 8;
                int col = nBase + nW + 16 * j + nl;
                float v = acc[i][j][r];
                if (bias)     v += bias[col];
                if (residual) v += residual[(size_t)row * N + col];
                if (relu)     v = fmaxf(v, 0.0f);
                if (outF) outF[(size_t)row * N + col] = v;
                else      outH[(size_t)row * N + col] = (hf)v;
            }
}

// ---------------------------------------------------------------------------
// Flash attention: head_dim=256, key blocks of 32, 8 waves x 16 queries each.
// grid = (S/128, B*H).  q/k/v/ctx are [B*S, 4096] f16, head h at cols h*256.
// K tile staged by async LDS copy; V transposed through registers.
// ---------------------------------------------------------------------------
__global__ __launch_bounds__(256) void attention_f16(const hf* __restrict__ q,
                                                     const hf* __restrict__ k,
                                                     const hf* __restrict__ v,
                                                     hf* __restrict__ ctx,
                                                     int S, int HD)
{
    __shared__ hf Ks[32][264];      // [key][dh]        row-major
    __shared__ hf Vt[256][40];      // [dh][key]        transposed
    __shared__ hf Pb[8][16][40];    // per-wave P tile (16x32)

    int t = threadIdx.x, lane = t & 31, wave = t >> 5;
    int bh = blockIdx.y;
    int b  = bh >> 4, h = bh & 15;
    size_t rowBase = (size_t)b * S;
    int col0 = h * 256;
    int q0 = blockIdx.x * 128 + wave * 16;

    const hf* qp = q + (rowBase + q0) * HD + col0;

    v16h qf[8];
#pragma unroll
    for (int kk = 0; kk < 8; ++kk)
        qf[kk] = load_a_frag(qp, HD, 0, kk * 32, lane);

    v8f o[16] = {};
    float mrow[8], lrow[8];
#pragma unroll
    for (int r = 0; r < 8; ++r) { mrow[r] = -3.0e38f; lrow[r] = 0.0f; }

    const float scale = 0.0625f;    // 1/sqrt(256)
    int hi = lane >> 4, nl = lane & 15;

    for (int j0 = 0; j0 < S; j0 += 32) {
        __syncthreads();            // prior iteration's Ks/Vt reads done
        {   // cooperative K/V tile load: thread -> 32 contiguous f16 of one key row
            int r = t >> 3;                 // 0..31 (key)
            int c = (t & 7) * 32;           // 0..224 (dh chunk)
            const hf* kp = k + (rowBase + j0 + r) * HD + col0 + c;
            const hf* vp = v + (rowBase + j0 + r) * HD + col0 + c;
#if HAVE_ASYNC_LDS
#pragma unroll
            for (int u = 0; u < 4; ++u)
                async_b128(kp + u * 8, &Ks[r][c + u * 8]);
#else
            h8 k0v = *(const h8*)kp;        h8 k1v = *(const h8*)(kp + 8);
            h8 k2v = *(const h8*)(kp + 16); h8 k3v = *(const h8*)(kp + 24);
            *(h8*)&Ks[r][c]      = k0v;     *(h8*)&Ks[r][c + 8]  = k1v;
            *(h8*)&Ks[r][c + 16] = k2v;     *(h8*)&Ks[r][c + 24] = k3v;
#endif
            h8 v0 = *(const h8*)vp;         h8 v1 = *(const h8*)(vp + 8);
            h8 v2 = *(const h8*)(vp + 16);  h8 v3 = *(const h8*)(vp + 24);
#pragma unroll
            for (int i = 0; i < 8; ++i) {
                Vt[c + i][r]      = v0[i];
                Vt[c + 8 + i][r]  = v1[i];
                Vt[c + 16 + i][r] = v2[i];
                Vt[c + 24 + i][r] = v3[i];
            }
        }
#if HAVE_ASYNC_LDS
        wait_async0();
#endif
        __syncthreads();

        // S = Q * K^T  (two 16x16 n-tiles over 8 k-steps)
        v8f sA[2] = {};
#pragma unroll
        for (int kk = 0; kk < 8; ++kk) {
            v16h kb0 = load_bt_frag(&Ks[0][0], 264, 0,  kk * 32, lane);
            v16h kb1 = load_bt_frag(&Ks[0][0], 264, 16, kk * 32, lane);
            sA[0] = wmma16(qf[kk], kb0, sA[0]);
            sA[1] = wmma16(qf[kk], kb1, sA[1]);
        }

        // online softmax; write P (f16) into LDS in C-layout
#pragma unroll
        for (int r = 0; r < 8; ++r) {
            float s0 = sA[0][r] * scale;
            float s1 = sA[1][r] * scale;
            float mx = fmaxf(s0, s1);
#pragma unroll
            for (int off = 8; off > 0; off >>= 1)
                mx = fmaxf(mx, __shfl_xor(mx, off, 32));
            float mnew = fmaxf(mrow[r], mx);
            float corr = __expf(mrow[r] - mnew);
            float e0 = __expf(s0 - mnew);
            float e1 = __expf(s1 - mnew);
            float rs = e0 + e1;
#pragma unroll
            for (int off = 8; off > 0; off >>= 1)
                rs += __shfl_xor(rs, off, 32);
            lrow[r] = lrow[r] * corr + rs;
            mrow[r] = mnew;
#pragma unroll
            for (int tt = 0; tt < 16; ++tt) o[tt][r] *= corr;
            Pb[wave][r + hi * 8][nl]      = (hf)e0;
            Pb[wave][r + hi * 8][16 + nl] = (hf)e1;
        }
        __syncthreads();

        // O += P * V
        v16h pf = load_a_frag(&Pb[wave][0][0], 40, 0, 0, lane);
#pragma unroll
        for (int tt = 0; tt < 16; ++tt) {
            v16h vb = load_bt_frag(&Vt[0][0], 40, tt * 16, 0, lane);
            o[tt] = wmma16(pf, vb, o[tt]);
        }
    }

    hf* cp = ctx + (rowBase + q0) * HD + col0;
#pragma unroll
    for (int r = 0; r < 8; ++r) {
        float linv = 1.0f / lrow[r];
#pragma unroll
        for (int tt = 0; tt < 16; ++tt)
            cp[(size_t)(r + hi * 8) * HD + tt * 16 + nl] = (hf)(o[tt][r] * linv);
    }
}

// ---------------------------------------------------------------------------
extern "C" void kernel_launch(void* const* d_in, const int* in_sizes, int n_in,
                              void* d_out, int out_size, void* d_ws, size_t ws_size,
                              hipStream_t stream)
{
    (void)in_sizes; (void)n_in; (void)out_size; (void)ws_size;
    const float* x   = (const float*)d_in[0];
    const float* Wq  = (const float*)d_in[1];
    const float* Wk  = (const float*)d_in[2];
    const float* Wv  = (const float*)d_in[3];
    const float* Wo  = (const float*)d_in[4];
    const float* W1  = (const float*)d_in[5];
    const float* b1  = (const float*)d_in[6];
    const float* W2  = (const float*)d_in[7];
    const float* b2  = (const float*)d_in[8];
    const float* g1  = (const float*)d_in[9];
    const float* be1 = (const float*)d_in[10];
    const float* g2  = (const float*)d_in[11];
    const float* be2 = (const float*)d_in[12];

    const int B = 2, S = 2048, D = 1024, DH = 4096, M = B * S;

    char* ws = (char*)d_ws;
    size_t off = 0;
    auto take = [&](size_t bytes) -> char* {
        char* p = ws + off;
        off += (bytes + 255) & ~(size_t)255;
        return p;
    };
    hf* WqT = (hf*)take((size_t)DH * D * 2);
    hf* WkT = (hf*)take((size_t)DH * D * 2);
    hf* WvT = (hf*)take((size_t)DH * D * 2);
    hf* W1T = (hf*)take((size_t)DH * D * 2);
    hf* WoT = (hf*)take((size_t)D * DH * 2);
    hf* W2T = (hf*)take((size_t)D * DH * 2);
    hf* xn1 = (hf*)take((size_t)M * D * 2);
    hf* qb  = (hf*)take((size_t)M * DH * 2);
    hf* kb  = (hf*)take((size_t)M * DH * 2);
    hf* vb  = (hf*)take((size_t)M * DH * 2);
    hf* cxb = (hf*)take((size_t)M * DH * 2);
    float* ybuf = (float*)take((size_t)M * D * 4);
    hf* xn2 = (hf*)take((size_t)M * D * 2);
    hf* hbf = (hf*)take((size_t)M * DH * 2);

    dim3 tb(32, 8);
    transpose_cast_f16<<<dim3(DH / 32, D / 32), tb, 0, stream>>>(Wq, WqT, D, DH);
    transpose_cast_f16<<<dim3(DH / 32, D / 32), tb, 0, stream>>>(Wk, WkT, D, DH);
    transpose_cast_f16<<<dim3(DH / 32, D / 32), tb, 0, stream>>>(Wv, WvT, D, DH);
    transpose_cast_f16<<<dim3(DH / 32, D / 32), tb, 0, stream>>>(W1, W1T, D, DH);
    transpose_cast_f16<<<dim3(D / 32, DH / 32), tb, 0, stream>>>(Wo, WoT, DH, D);
    transpose_cast_f16<<<dim3(D / 32, DH / 32), tb, 0, stream>>>(W2, W2T, DH, D);

    layernorm_f16<<<M, 256, 0, stream>>>(x, g1, be1, xn1, D);

    gemm_f16<<<dim3(DH / 128, M / 128), 256, 0, stream>>>(xn1, WqT, M, DH, D,
                                                          nullptr, nullptr, 0, nullptr, qb);
    gemm_f16<<<dim3(DH / 128, M / 128), 256, 0, stream>>>(xn1, WkT, M, DH, D,
                                                          nullptr, nullptr, 0, nullptr, kb);
    gemm_f16<<<dim3(DH / 128, M / 128), 256, 0, stream>>>(xn1, WvT, M, DH, D,
                                                          nullptr, nullptr, 0, nullptr, vb);

    attention_f16<<<dim3(S / 128, B * 16), 256, 0, stream>>>(qb, kb, vb, cxb, S, DH);

    // y = x + ctx @ Wo
    gemm_f16<<<dim3(D / 128, M / 128), 256, 0, stream>>>(cxb, WoT, M, D, DH,
                                                         nullptr, x, 0, ybuf, nullptr);

    layernorm_f16<<<M, 256, 0, stream>>>(ybuf, g2, be2, xn2, D);

    // h = relu(xn2 @ W1 + b1)
    gemm_f16<<<dim3(DH / 128, M / 128), 256, 0, stream>>>(xn2, W1T, M, DH, D,
                                                          b1, nullptr, 1, nullptr, hbf);
    // out = y + h @ W2 + b2
    gemm_f16<<<dim3(D / 128, M / 128), 256, 0, stream>>>(hbf, W2T, M, D, DH,
                                                         b2, ybuf, 0, (float*)d_out, nullptr);
}
